// LSTMPcell_48017734369974
// MI455X (gfx1250) — matrix operand order repaired and checked by reading, not compiled, not run on hardware
//
#include <hip/hip_runtime.h>

typedef unsigned short u16;
typedef __attribute__((ext_vector_type(16))) __bf16 v16bf;
typedef __attribute__((ext_vector_type(8)))  float  v8f;

#define BATCH   16384
#define HID     512
#define KDIM    1024   // I + H
#define NGATES  2048   // 4 * H

// ---------------------------------------------------------------- helpers ---
union FragU { uint4 u[2]; v16bf v; };

__device__ __forceinline__ u16 f2bf(float x) {
  unsigned u = __float_as_uint(x);
  u += 0x7FFFu + ((u >> 16) & 1u);      // round-to-nearest-even
  return (u16)(u >> 16);
}

__device__ __forceinline__ float sigmoid_f(float x) {
  return 1.0f / (1.0f + __expf(-x));
}

__device__ __forceinline__ float tanh_f(float x) {
  x = fminf(fmaxf(x, -15.0f), 15.0f);
  float e = __expf(2.0f * x);
  return (e - 1.0f) / (e + 1.0f);
}

// A fragment (16x32 bf16, MxK): lanes 0-15 hold row M=lane, K {0..7,16..23};
// lanes 16-31 hold row M=lane-16, K {8..15,24..31}. Row-major source.
__device__ __forceinline__ v16bf load_a_frag(const u16* __restrict__ A, int lda,
                                             int m_base, int k0, int lane) {
  int row  = m_base + (lane & 15);
  int koff = (lane & 16) ? 8 : 0;
  const u16* p = A + (size_t)row * lda + k0 + koff;
  FragU f;
  f.u[0] = *(const uint4*)(p);        // K = koff .. koff+7
  f.u[1] = *(const uint4*)(p + 16);   // K = koff+16 .. koff+23
  return f.v;
}

// B fragment (32x16 bf16, KxN) for D = A * W^T with W [N,K] row-major:
// lane holds column n = n_base + (lane&15), K values from row n of W
// (lanes 0-15: K {0..7,16..23}, lanes 16-31: K {8..15,24..31}).
__device__ __forceinline__ v16bf load_b_frag(const u16* __restrict__ W, int ldw,
                                             int n_base, int k0, int lane) {
  int col = n_base + (lane & 15);
  int kk  = k0 + ((lane & 16) ? 8 : 0);
  const u16* p = W + (size_t)col * ldw + kk;
  FragU f;
  f.u[0] = *(const uint4*)(p);        // K = kk .. kk+7
  f.u[1] = *(const uint4*)(p + 16);   // K = kk+16 .. kk+23
  return f.v;
}

__device__ __forceinline__ v8f v8f_zero() {
  v8f z = {0.f, 0.f, 0.f, 0.f, 0.f, 0.f, 0.f, 0.f};
  return z;
}

// ----------------------------------------------------- bf16 pack kernels ----
__global__ __launch_bounds__(256) void pack_a_kernel(
    const float* __restrict__ x, const float* __restrict__ h,
    u16* __restrict__ Acat) {
  size_t i = (size_t)blockIdx.x * 256 + threadIdx.x;   // over BATCH*KDIM
  int row = (int)(i >> 10);
  int col = (int)(i & 1023);
  float v = (col < HID) ? x[(size_t)row * HID + col]
                        : h[(size_t)row * HID + (col - HID)];
  Acat[i] = f2bf(v);
}

__global__ __launch_bounds__(256) void cvt_bf16_kernel(
    const float* __restrict__ src, u16* __restrict__ dst, int n) {
  int i = blockIdx.x * 256 + threadIdx.x;
  if (i < n) dst[i] = f2bf(src[i]);
}

// -------------------------------------- fused gates + cell + layernorm ------
// Block: 512 threads = 16 waves, arranged 2 (M) x 8 (N).
// Tile: 32 rows x 2048 gate columns, K = 1024 in steps of 32.
// Dynamic LDS: gates[32][2048] fp32 = 256 KB (<= 320 KB WGP LDS).
__global__ __launch_bounds__(512) void lstm_gates_kernel(
    const u16*  __restrict__ Acat,   // [BATCH, KDIM] bf16
    const u16*  __restrict__ Wbf,    // [NGATES, KDIM] bf16
    const float* __restrict__ bias,  // [NGATES]
    const float* __restrict__ c_prev,// [BATCH, HID]
    const float* __restrict__ gamma, // [HID]
    const float* __restrict__ beta,  // [HID]
    float* __restrict__ cell_out,    // [BATCH, HID] fp32 (output #2)
    u16*  __restrict__ act_out)      // [BATCH, HID] bf16 (proj input)
{
  extern __shared__ float gates[];   // 32 * 2048 floats

  const int tid  = threadIdx.x;
  const int wave = tid >> 5;
  const int lane = tid & 31;
  const int mw   = wave >> 3;        // 0..1  (M wave)
  const int nw   = wave & 7;         // 0..7  (N wave)
  const int m_base = blockIdx.x * 32;
  const int m_wave = m_base + mw * 16;
  const int n_wave = nw * 256;

  v8f acc[16];
#pragma unroll
  for (int t = 0; t < 16; ++t) acc[t] = v8f_zero();

  for (int k0 = 0; k0 < KDIM; k0 += 32) {
    v16bf a = load_a_frag(Acat, KDIM, m_wave, k0, lane);
#pragma unroll
    for (int t = 0; t < 16; ++t) {
      v16bf b = load_b_frag(Wbf, KDIM, n_wave + t * 16, k0, lane);
      acc[t] = __builtin_amdgcn_wmma_f32_16x16x32_bf16(
          false, a, false, b, (short)0, acc[t], false, false);
    }
  }

  // D layout: VGPR j, lanes 0-15 -> (row j, col lane); lanes 16-31 -> (row j+8).
  const int colL  = lane & 15;
  const int rhalf = (lane & 16) ? 8 : 0;
#pragma unroll
  for (int t = 0; t < 16; ++t) {
    int col  = n_wave + t * 16 + colL;
    float bv = bias[col];
#pragma unroll
    for (int j = 0; j < 8; ++j)
      gates[(size_t)(mw * 16 + rhalf + j) * NGATES + col] = acc[t][j] + bv;
  }
  __syncthreads();

  // Elementwise + LayerNorm: wave w handles rows 2w and 2w+1 of the tile.
#pragma unroll
  for (int rr = 0; rr < 2; ++rr) {
    const int r    = wave * 2 + rr;
    const int grow = m_base + r;
    const float* grow_gates = &gates[(size_t)r * NGATES];

    float cv_reg[16];
    float sum = 0.f, sumsq = 0.f;
#pragma unroll
    for (int idx = 0; idx < 16; ++idx) {
      int h = lane + idx * 32;
      float iv = sigmoid_f(grow_gates[h]);
      float fv = sigmoid_f(grow_gates[512 + h]);
      float gv = tanh_f(grow_gates[1024 + h]);
      float cv = fv * c_prev[(size_t)grow * HID + h] + iv * gv;
      cv_reg[idx] = cv;
      sum   += cv;
      sumsq += cv * cv;
    }
#pragma unroll
    for (int off = 16; off > 0; off >>= 1) {   // wave32 butterfly reduce
      sum   += __shfl_xor(sum,   off, 32);
      sumsq += __shfl_xor(sumsq, off, 32);
    }
    float mu   = sum * (1.0f / 512.0f);
    float var  = sumsq * (1.0f / 512.0f) - mu * mu;
    float rstd = rsqrtf(var + 1e-5f);
#pragma unroll
    for (int idx = 0; idx < 16; ++idx) {
      int h = lane + idx * 32;
      float cn = (cv_reg[idx] - mu) * rstd * gamma[h] + beta[h];
      cell_out[(size_t)grow * HID + h] = cn;
      float ov = sigmoid_f(grow_gates[1536 + h]);
      act_out[(size_t)grow * HID + h] = f2bf(ov * tanh_f(cn));
    }
  }
}

// ----------------------------------------------- projection: out = act*Wp^T -
// Block: 256 threads = 8 waves, 1 (M) x 8 (N). Tile: 16 rows x 512 cols.
__global__ __launch_bounds__(256) void proj_kernel(
    const u16* __restrict__ act,   // [BATCH, HID] bf16
    const u16* __restrict__ Wpbf,  // [HID, HID] bf16
    float* __restrict__ out)       // [BATCH, HID] fp32 (output #1)
{
  const int wave = threadIdx.x >> 5;
  const int lane = threadIdx.x & 31;
  const int m_base = blockIdx.x * 16;
  const int n_wave = wave * 64;

  v8f acc[4];
#pragma unroll
  for (int t = 0; t < 4; ++t) acc[t] = v8f_zero();

  for (int k0 = 0; k0 < HID; k0 += 32) {
    v16bf a = load_a_frag(act, HID, m_base, k0, lane);
#pragma unroll
    for (int t = 0; t < 4; ++t) {
      v16bf b = load_b_frag(Wpbf, HID, n_wave + t * 16, k0, lane);
      acc[t] = __builtin_amdgcn_wmma_f32_16x16x32_bf16(
          false, a, false, b, (short)0, acc[t], false, false);
    }
  }

  const int colL  = lane & 15;
  const int rhalf = (lane & 16) ? 8 : 0;
#pragma unroll
  for (int t = 0; t < 4; ++t) {
    int col = n_wave + t * 16 + colL;
#pragma unroll
    for (int j = 0; j < 8; ++j)
      out[(size_t)(m_base + rhalf + j) * HID + col] = acc[t][j];
  }
}

// ------------------------------------------------------------------ launch --
extern "C" void kernel_launch(void* const* d_in, const int* in_sizes, int n_in,
                              void* d_out, int out_size, void* d_ws, size_t ws_size,
                              hipStream_t stream) {
  const float* x      = (const float*)d_in[0];
  const float* h_prev = (const float*)d_in[1];
  const float* c_prev = (const float*)d_in[2];
  const float* W      = (const float*)d_in[3];
  const float* b      = (const float*)d_in[4];
  const float* gamma  = (const float*)d_in[5];
  const float* beta   = (const float*)d_in[6];
  const float* Wp     = (const float*)d_in[7];

  float* out  = (float*)d_out;                       // [BATCH, HID]
  float* cell = out + (size_t)BATCH * HID;           // [BATCH, HID]

  char* ws = (char*)d_ws;
  const size_t szA  = (size_t)BATCH  * KDIM * sizeof(u16);   // 32 MB
  const size_t szW  = (size_t)NGATES * KDIM * sizeof(u16);   //  4 MB
  const size_t szWp = (size_t)HID    * HID  * sizeof(u16);   // 0.5 MB
  u16* Acat = (u16*)(ws);
  u16* Wbf  = (u16*)(ws + szA);
  u16* Wpbf = (u16*)(ws + szA + szW);
  u16* act  = (u16*)(ws + szA + szW + szWp);                 // 16 MB

  // 1) fp32 -> bf16 packing
  pack_a_kernel<<<(BATCH * KDIM) / 256, 256, 0, stream>>>(x, h_prev, Acat);
  cvt_bf16_kernel<<<(NGATES * KDIM) / 256, 256, 0, stream>>>(W, Wbf, NGATES * KDIM);
  cvt_bf16_kernel<<<(HID * HID) / 256, 256, 0, stream>>>(Wp, Wpbf, HID * HID);

  // 2) fused gates GEMM + cell update + LayerNorm (256 KB dynamic LDS)
  lstm_gates_kernel<<<BATCH / 32, 512, 32 * NGATES * sizeof(float), stream>>>(
      Acat, Wbf, b, c_prev, gamma, beta, cell, act);

  // 3) projection GEMM
  proj_kernel<<<BATCH / 16, 256, 0, stream>>>(act, Wpbf, out);
}
